// MPNNet_44220983279845
// MI455X (gfx1250) — matrix operand order repaired and checked by reading, not compiled, not run on hardware
//
#include <hip/hip_runtime.h>
#include <hip/hip_bf16.h>
#include <cstdint>

typedef __attribute__((ext_vector_type(16))) _Float16 v16h;
typedef __attribute__((ext_vector_type(8)))  float    v8f;
typedef __attribute__((ext_vector_type(4)))  int      v4i;

#define NNODES 50000
#define NEDGES 100000
#define EBLK   64
#define NEBLK  ((NEDGES + EBLK - 1) / EBLK)   // 1563
#define EPAD   (NEBLK * EBLK)                 // 100032
#define NGRAPH 2048
#define NFEAT  14
#define DD     64

union Frag  { v16h h; v4i q[2]; unsigned u[8]; };
union HBits { _Float16 f; unsigned short u; };

__device__ __forceinline__ v8f wmma16(const Frag& a, const Frag& b, v8f c) {
  return __builtin_amdgcn_wmma_f32_16x16x32_f16(false, a.h, false, b.h, (short)0, c, false, false);
}
__device__ __forceinline__ float sigm(float x) { return 1.f / (1.f + expf(-x)); }

// ---------------- utility ----------------
__global__ void k_zero(float* __restrict__ p, long long n) {
  long long i = (long long)blockIdx.x * 256 + threadIdx.x;
  if (i < n) p[i] = 0.f;
}

__global__ void k_deg(const int* __restrict__ dst, float* __restrict__ deg) {
  int e = blockIdx.x * 256 + threadIdx.x;
  if (e < NEDGES) atomicAdd(&deg[dst[e]], 1.f);
}
__global__ void k_invdeg(const float* __restrict__ deg, float* __restrict__ inv) {
  int n = blockIdx.x * 256 + threadIdx.x;
  if (n < NNODES) inv[n] = 1.f / fmaxf(deg[n], 1.f);
}

// out = relu(x @ lin0_w.T + b)  (K=14, tiny -> VALU)
__global__ void k_lin0(const float* __restrict__ x, const float* __restrict__ w,
                       const float* __restrict__ b, float* __restrict__ out,
                       _Float16* __restrict__ out16) {
  int idx = blockIdx.x * 256 + threadIdx.x;
  if (idx >= NNODES * DD) return;
  int n = idx >> 6, d = idx & 63;
  float acc = b[d];
#pragma unroll
  for (int f = 0; f < NFEAT; ++f) acc += x[n * NFEAT + f] * w[d * NFEAT + f];
  acc = fmaxf(acc, 0.f);
  out[idx] = acc;
  out16[idx] = (_Float16)acc;
}

// h = relu(edge_attr @ nn_w1.T + b1), f16 output [E,128]
__global__ void k_edgeh(const float* __restrict__ ea, const float* __restrict__ w1,
                        const float* __restrict__ b1, _Float16* __restrict__ h) {
  int idx = blockIdx.x * 256 + threadIdx.x;
  if (idx >= NEDGES * 128) return;
  int e = idx >> 7, k = idx & 127;
  float acc = b1[k];
#pragma unroll
  for (int j = 0; j < 4; ++j) acc += ea[e * 4 + j] * w1[k * 4 + j];
  h[idx] = (_Float16)fmaxf(acc, 0.f);
}

// ---------------- B-operand packing (WMMA 16x16x32 f16 layout) ----------------
// dst[((ks*ntiles + nt)*32 + lane)*8 + j] packs B[k0,n],B[k0+1,n]
// with n = nt*16 + (lane&15), k0 = ks*32 + 8*(lane>=16) + 16*(j>=4) + 2*(j&3)
__global__ void k_pack_b(const float* __restrict__ src, unsigned* __restrict__ dst,
                         int K, int N, int sk, int sn) {
  int idx = blockIdx.x * 256 + threadIdx.x;
  int ntiles = N >> 4;
  int total = (K >> 5) * ntiles * 32 * 8;
  if (idx >= total) return;
  int j = idx & 7, lane = (idx >> 3) & 31, rest = idx >> 8;
  int nt = rest % ntiles, ks = rest / ntiles;
  int n = nt * 16 + (lane & 15);
  int k0 = ks * 32 + ((lane >> 4) << 3) + ((j & 4) << 2) + 2 * (j & 3);
  HBits h0, h1;
  h0.f = (_Float16)src[(size_t)k0 * sk + (size_t)n * sn];
  h1.f = (_Float16)src[(size_t)(k0 + 1) * sk + (size_t)n * sn];
  dst[idx] = (unsigned)h0.u | ((unsigned)h1.u << 16);
}

// W2p[i][k][o] = nn_w2[(i*64+o)*128 + k], packed per (i, ks, nt)
__global__ void k_pack_edge(const float* __restrict__ w2, unsigned* __restrict__ dst) {
  int idx = blockIdx.x * 256 + threadIdx.x;
  if (idx >= 64 * 4 * 4 * 32 * 8) return;
  int j = idx & 7, lane = (idx >> 3) & 31, rest = idx >> 8;
  int nt = rest & 3; rest >>= 2;
  int ks = rest & 3; int i = rest >> 2;
  int o = nt * 16 + (lane & 15);
  int k0 = ks * 32 + ((lane >> 4) << 3) + ((j & 4) << 2) + 2 * (j & 3);
  size_t rbase = (size_t)(i * 64 + o) * 128;
  HBits h0, h1;
  h0.f = (_Float16)w2[rbase + k0];
  h1.f = (_Float16)w2[rbase + k0 + 1];
  dst[idx] = (unsigned)h0.u | ((unsigned)h1.u << 16);
}

// ---------------- fused edge-message WMMA kernel ----------------
// msg[e,o] = sum_i out16[src_e,i] * ( sum_k h[e,k]*W2p[i,k,o] )  +  out16[src_e,:] @ B2
// block: 64 edges x 64 outs; 8 waves = 2 edge-groups x 4 n-tiles;
// each wave keeps 2 accumulators (2 edge-subtiles) so every B fragment is reused
// twice -> halves L2 B-traffic (arith intensity ~32 FLOP per L2 byte).
__global__ __launch_bounds__(256) void k_edge_msg(
    const _Float16* __restrict__ h_edge, const _Float16* __restrict__ out16,
    const int* __restrict__ src, const unsigned* __restrict__ bpack,
    const unsigned* __restrict__ b2pack, float* __restrict__ msg) {
  __shared__ __align__(16) _Float16 sh[EBLK][128];  // h tile, 16KB
  __shared__ __align__(16) _Float16 sa[EBLK][64];   // gathered out[src], 8KB
  int tid = threadIdx.x;
  int ebase = blockIdx.x * EBLK;

  unsigned* shw = (unsigned*)&sh[0][0];
  for (int i = tid; i < EBLK * 64; i += 256) {      // 64 dwords per h row
    int row = i >> 6, dw = i & 63;
    int e = ebase + row;
    shw[i] = (e < NEDGES) ? ((const unsigned*)h_edge)[(size_t)e * 64 + dw] : 0u;
  }
  unsigned* saw = (unsigned*)&sa[0][0];
  for (int i = tid; i < EBLK * 32; i += 256) {      // 32 dwords per out row
    int row = i >> 5, dw = i & 31;
    int e = ebase + row;
    int s = (e < NEDGES) ? src[e] : 0;
    saw[i] = ((const unsigned*)(out16 + (size_t)s * 64))[dw];
  }
  __syncthreads();

  int wave = tid >> 5, lane = tid & 31, hi = lane >> 4;
  int eg = wave >> 2, nt = wave & 3;
  int r0 = eg * 32 + (lane & 15);                   // rows r0 (acc0) and r0+16 (acc1)
  const unsigned* hrow0 = (const unsigned*)&sh[r0][0];
  const unsigned* hrow1 = (const unsigned*)&sh[r0 + 16][0];
  const _Float16* arow0 = &sa[r0][0];
  const _Float16* arow1 = &sa[r0 + 16][0];
  const unsigned* arow0d = (const unsigned*)arow0;
  const unsigned* arow1d = (const unsigned*)arow1;

  v8f acc0 = {}, acc1 = {};
  for (int i = 0; i < 64; ++i) {
    _Float16 s0 = arow0[i];
    _Float16 s1 = arow1[i];
#pragma unroll
    for (int ks = 0; ks < 4; ++ks) {
      Frag b;
      const unsigned* bp = bpack + ((((size_t)i * 4 + ks) * 4 + nt) * 32 + lane) * 8;
      b.q[0] = *(const v4i*)(bp);
      b.q[1] = *(const v4i*)(bp + 4);

      Frag a0;
      const unsigned* hp0 = hrow0 + ks * 16 + hi * 4;
      a0.q[0] = *(const v4i*)(hp0);
      a0.q[1] = *(const v4i*)(hp0 + 8);
#pragma unroll
      for (int q = 0; q < 16; ++q) a0.h[q] = a0.h[q] * s0;
      acc0 = wmma16(a0, b, acc0);

      Frag a1;
      const unsigned* hp1 = hrow1 + ks * 16 + hi * 4;
      a1.q[0] = *(const v4i*)(hp1);
      a1.q[1] = *(const v4i*)(hp1 + 8);
#pragma unroll
      for (int q = 0; q < 16; ++q) a1.h[q] = a1.h[q] * s1;
      acc1 = wmma16(a1, b, acc1);
    }
  }
  // bias contribution: out_src @ B2   (K=64 -> 2 k-steps), B reused for both accs
#pragma unroll
  for (int ks = 0; ks < 2; ++ks) {
    Frag b;
    const unsigned* bp = b2pack + (((size_t)ks * 4 + nt) * 32 + lane) * 8;
    b.q[0] = *(const v4i*)(bp);
    b.q[1] = *(const v4i*)(bp + 4);

    Frag a0;
    const unsigned* ap0 = arow0d + ks * 16 + hi * 4;
    a0.q[0] = *(const v4i*)(ap0);
    a0.q[1] = *(const v4i*)(ap0 + 8);
    acc0 = wmma16(a0, b, acc0);

    Frag a1;
    const unsigned* ap1 = arow1d + ks * 16 + hi * 4;
    a1.q[0] = *(const v4i*)(ap1);
    a1.q[1] = *(const v4i*)(ap1 + 8);
    acc1 = wmma16(a1, b, acc1);
  }
  int col = nt * 16 + (lane & 15);
  int rb0 = ebase + eg * 32 + hi * 8;
#pragma unroll
  for (int j = 0; j < 8; ++j) {
    msg[(size_t)(rb0 + j) * 64 + col] = acc0[j];          // rows [eg*32 .. +15]
    msg[(size_t)(rb0 + 16 + j) * 64 + col] = acc1[j];     // rows [eg*32+16 .. +31]
  }
}

// ---------------- generic node WMMA GEMM ----------------
// C[M,N] = act( A_f16[M,K] @ Bpack + bias[N] + addend[M,N]*rscale[M] ), optional f16 mirror
__global__ __launch_bounds__(256) void k_gemm(
    const _Float16* __restrict__ A, const unsigned* __restrict__ bpack,
    const float* __restrict__ bias, const float* __restrict__ addend,
    const float* __restrict__ rscale, float* __restrict__ C,
    _Float16* __restrict__ C16, int M, int N, int K, int do_relu) {
  int ntiles = N >> 4;
  int wid = blockIdx.x * 8 + (threadIdx.x >> 5);
  int mt = wid / ntiles, nt = wid - mt * ntiles;
  if (mt * 16 >= M) return;  // wave-uniform
  int lane = threadIdx.x & 31, hi = lane >> 4;
  const unsigned* arow = (const unsigned*)(A + (size_t)(mt * 16 + (lane & 15)) * K);
  v8f acc = {};
  int ksteps = K >> 5;
  for (int ks = 0; ks < ksteps; ++ks) {
    Frag a, b;
    const unsigned* ap = arow + ks * 16 + hi * 4;
    a.q[0] = *(const v4i*)(ap);
    a.q[1] = *(const v4i*)(ap + 8);
    const unsigned* bp = bpack + (((size_t)ks * ntiles + nt) * 32 + lane) * 8;
    b.q[0] = *(const v4i*)(bp);
    b.q[1] = *(const v4i*)(bp + 4);
    acc = wmma16(a, b, acc);
  }
  int col = nt * 16 + (lane & 15);
#pragma unroll
  for (int j = 0; j < 8; ++j) {
    int row = mt * 16 + hi * 8 + j;
    float v = acc[j] + bias[col];
    if (addend) {
      float s = rscale ? rscale[row] : 1.f;
      v += addend[(size_t)row * N + col] * s;
    }
    if (do_relu) v = fmaxf(v, 0.f);
    C[(size_t)row * N + col] = v;
    if (C16) C16[(size_t)row * N + col] = (_Float16)v;
  }
}

// agg[dst[e],:] += msg[e,:]
__global__ void k_scatter(const float* __restrict__ msg, const int* __restrict__ dst,
                          float* __restrict__ agg) {
  int idx = blockIdx.x * 256 + threadIdx.x;
  if (idx >= NEDGES * DD) return;
  int e = idx >> 6, d = idx & 63;
  atomicAdd(&agg[(size_t)dst[e] * DD + d], msg[idx]);
}

// GRU elementwise: out <- (1-z)*n + z*out
__global__ void k_gru(const float* __restrict__ g1, const float* __restrict__ g2,
                      float* __restrict__ out, _Float16* __restrict__ out16) {
  int idx = blockIdx.x * 256 + threadIdx.x;
  if (idx >= NNODES * DD) return;
  int n = idx >> 6, d = idx & 63;
  size_t b = (size_t)n * 192;
  float r  = sigm(g1[b + d] + g2[b + d]);
  float z  = sigm(g1[b + 64 + d] + g2[b + 64 + d]);
  float nn = tanhf(g1[b + 128 + d] + r * g2[b + 128 + d]);
  float h  = out[idx];
  float hn = (1.f - z) * nn + z * h;
  out[idx] = hn;
  out16[idx] = (_Float16)hn;
}

// ---------------- Set2Set ----------------
__global__ void k_lstm(const float* __restrict__ qstar, const float* __restrict__ hhp,
                       float* __restrict__ cc, float* __restrict__ hhn,
                       const float* __restrict__ wih, const float* __restrict__ whh,
                       const float* __restrict__ bih, const float* __restrict__ bhh) {
  int idx = blockIdx.x * 256 + threadIdx.x;
  if (idx >= NGRAPH * DD) return;
  int bg = idx >> 6, d = idx & 63;
  float g[4];
#pragma unroll
  for (int gg = 0; gg < 4; ++gg) {
    int j = gg * 64 + d;
    float acc = bih[j] + bhh[j];
    const float* q  = qstar + (size_t)bg * 128;
    const float* wr = wih + (size_t)j * 128;
    for (int k = 0; k < 128; ++k) acc += q[k] * wr[k];
    const float* hp  = hhp + (size_t)bg * 64;
    const float* wr2 = whh + (size_t)j * 64;
    for (int k = 0; k < 64; ++k) acc += hp[k] * wr2[k];
    g[gg] = acc;
  }
  float i_ = sigm(g[0]), f_ = sigm(g[1]), gv = tanhf(g[2]), o_ = sigm(g[3]);
  float c2 = f_ * cc[idx] + i_ * gv;
  cc[idx] = c2;
  hhn[idx] = o_ * tanhf(c2);
}

__device__ __forceinline__ unsigned fenc(float x) {
  unsigned b = __float_as_uint(x);
  return (b & 0x80000000u) ? ~b : (b | 0x80000000u);
}
__device__ __forceinline__ float fdec(unsigned u) {
  unsigned b = (u & 0x80000000u) ? (u & 0x7fffffffu) : ~u;
  return __uint_as_float(b);
}

__global__ void k_attdot(const float* __restrict__ out, const float* __restrict__ hh,
                         const int* __restrict__ batch, float* __restrict__ e) {
  int n = blockIdx.x * 256 + threadIdx.x;
  if (n >= NNODES) return;
  const float* o = out + (size_t)n * 64;
  const float* h = hh + (size_t)batch[n] * 64;
  float acc = 0.f;
#pragma unroll
  for (int d = 0; d < 64; ++d) acc += o[d] * h[d];
  e[n] = acc;
}
__global__ void k_segmax(const float* __restrict__ e, const int* __restrict__ batch,
                         unsigned* __restrict__ mx) {
  int n = blockIdx.x * 256 + threadIdx.x;
  if (n < NNODES) atomicMax(&mx[batch[n]], fenc(e[n]));
}
__global__ void k_atta(const float* __restrict__ e, const int* __restrict__ batch,
                       const unsigned* __restrict__ mx, float* __restrict__ a,
                       float* __restrict__ den) {
  int n = blockIdx.x * 256 + threadIdx.x;
  if (n >= NNODES) return;
  int g = batch[n];
  float av = expf(e[n] - fdec(mx[g]));
  a[n] = av;
  atomicAdd(&den[g], av);
}
__global__ void k_attw(const float* __restrict__ a, const int* __restrict__ batch,
                       const float* __restrict__ den, float* __restrict__ w) {
  int n = blockIdx.x * 256 + threadIdx.x;
  if (n < NNODES) w[n] = a[n] / (den[batch[n]] + 1e-16f);
}
__global__ void k_attr(const float* __restrict__ w, const float* __restrict__ out,
                       const int* __restrict__ batch, float* __restrict__ r) {
  int idx = blockIdx.x * 256 + threadIdx.x;
  if (idx >= NNODES * DD) return;
  int n = idx >> 6, d = idx & 63;
  atomicAdd(&r[(size_t)batch[n] * 64 + d], w[n] * out[idx]);
}
__global__ void k_qstar(const float* __restrict__ hh, const float* __restrict__ r,
                        float* __restrict__ q) {
  int idx = blockIdx.x * 256 + threadIdx.x;
  if (idx >= NGRAPH * 128) return;
  int b = idx >> 7, d = idx & 127;
  q[idx] = (d < 64) ? hh[(size_t)b * 64 + d] : r[(size_t)b * 64 + d - 64];
}

__global__ void k_final(const float* __restrict__ q, const float* __restrict__ w1,
                        const float* __restrict__ b1, const float* __restrict__ w2,
                        const float* __restrict__ b2, float* __restrict__ out) {
  int b = blockIdx.x * 256 + threadIdx.x;
  if (b >= NGRAPH) return;
  const float* qb = q + (size_t)b * 128;
  float acc = b2[0];
  for (int d = 0; d < 64; ++d) {
    float y = b1[d];
    const float* wr = w1 + (size_t)d * 128;
    for (int k = 0; k < 128; ++k) y += qb[k] * wr[k];
    acc += fmaxf(y, 0.f) * w2[d];
  }
  out[b] = acc;
}

// ---------------- host ----------------
#define LB(n) dim3((unsigned)(((n) + 255) / 256)), dim3(256), 0, stream

extern "C" void kernel_launch(void* const* d_in, const int* in_sizes, int n_in,
                              void* d_out, int out_size, void* d_ws, size_t ws_size,
                              hipStream_t stream) {
  (void)in_sizes; (void)n_in; (void)out_size; (void)ws_size;
  const float* x         = (const float*)d_in[0];
  const int*   ei        = (const int*)  d_in[1];
  const float* ea        = (const float*)d_in[2];
  const int*   batch     = (const int*)  d_in[3];
  const float* lin0_w    = (const float*)d_in[4];
  const float* lin0_b    = (const float*)d_in[5];
  const float* nn_w1     = (const float*)d_in[6];
  const float* nn_b1     = (const float*)d_in[7];
  const float* nn_w2     = (const float*)d_in[8];
  const float* nn_b2     = (const float*)d_in[9];
  const float* conv_root = (const float*)d_in[10];
  const float* conv_bias = (const float*)d_in[11];
  const float* gru_w_ih  = (const float*)d_in[12];
  const float* gru_w_hh  = (const float*)d_in[13];
  const float* gru_b_ih  = (const float*)d_in[14];
  const float* gru_b_hh  = (const float*)d_in[15];
  const float* s2s_w_ih  = (const float*)d_in[16];
  const float* s2s_w_hh  = (const float*)d_in[17];
  const float* s2s_b_ih  = (const float*)d_in[18];
  const float* s2s_b_hh  = (const float*)d_in[19];
  const float* lin1_w    = (const float*)d_in[20];
  const float* lin1_b    = (const float*)d_in[21];
  const float* lin2_w    = (const float*)d_in[22];
  const float* lin2_b    = (const float*)d_in[23];

  const int* src  = ei;            // edge_index[0]
  const int* dstE = ei + NEDGES;   // edge_index[1]

  char* W = (char*)d_ws;
  size_t off = 0;
  auto alloc = [&](size_t bytes) { size_t o = off; off += (bytes + 255) & ~(size_t)255; return o; };

  float*     deg   = (float*)    (W + alloc((size_t)NNODES * 4));
  float*     invd  = (float*)    (W + alloc((size_t)NNODES * 4));
  float*     out   = (float*)    (W + alloc((size_t)NNODES * 64 * 4));
  _Float16*  out16 = (_Float16*) (W + alloc((size_t)NNODES * 64 * 2));
  _Float16*  hedge = (_Float16*) (W + alloc((size_t)NEDGES * 128 * 2));
  unsigned*  bpE   = (unsigned*) (W + alloc((size_t)64 * 4 * 4 * 32 * 8 * 4)); // 1MB
  unsigned*  bp2   = (unsigned*) (W + alloc((size_t)2048 * 4));
  unsigned*  bpR   = (unsigned*) (W + alloc((size_t)2048 * 4));
  unsigned*  bpI   = (unsigned*) (W + alloc((size_t)6144 * 4));
  unsigned*  bpH   = (unsigned*) (W + alloc((size_t)6144 * 4));
  float*     msg   = (float*)    (W + alloc((size_t)EPAD * 64 * 4));  // padded rows
  float*     agg   = (float*)    (W + alloc((size_t)NNODES * 64 * 4));
  float*     mbuf  = (float*)    (W + alloc((size_t)NNODES * 64 * 4));
  _Float16*  m16   = (_Float16*) (W + alloc((size_t)NNODES * 64 * 2));
  float*     g1    = (float*)    (W + alloc((size_t)NNODES * 192 * 4));
  float*     g2    = (float*)    (W + alloc((size_t)NNODES * 192 * 4));
  float*     hh0   = (float*)    (W + alloc((size_t)NGRAPH * 64 * 4));
  float*     hh1   = (float*)    (W + alloc((size_t)NGRAPH * 64 * 4));
  float*     ccb   = (float*)    (W + alloc((size_t)NGRAPH * 64 * 4));
  float*     qst   = (float*)    (W + alloc((size_t)NGRAPH * 128 * 4));
  float*     eatt  = (float*)    (W + alloc((size_t)NNODES * 4));
  float*     aatt  = (float*)    (W + alloc((size_t)NNODES * 4));
  float*     watt  = (float*)    (W + alloc((size_t)NNODES * 4));
  unsigned*  mxb   = (unsigned*) (W + alloc((size_t)NGRAPH * 4));
  float*     den   = (float*)    (W + alloc((size_t)NGRAPH * 4));
  float*     rbuf  = (float*)    (W + alloc((size_t)NGRAPH * 64 * 4));

  // ---- prep ----
  k_zero<<<LB(NNODES)>>>(deg, NNODES);
  k_deg<<<LB(NEDGES)>>>(dstE, deg);
  k_invdeg<<<LB(NNODES)>>>(deg, invd);
  k_lin0<<<LB(NNODES * 64)>>>(x, lin0_w, lin0_b, out, out16);
  k_edgeh<<<LB(NEDGES * 128)>>>(ea, nn_w1, nn_b1, hedge);
  k_pack_edge<<<LB(64 * 4 * 4 * 32 * 8)>>>(nn_w2, bpE);
  k_pack_b<<<LB(2048)>>>(nn_b2,     bp2, 64, 64,  64, 1);   // B2[i,o]=b2[i*64+o]
  k_pack_b<<<LB(2048)>>>(conv_root, bpR, 64, 64,  64, 1);   // B[k,n]=conv_root[k,n]
  k_pack_b<<<LB(6144)>>>(gru_w_ih,  bpI, 64, 192, 1,  64);  // B[k,n]=w_ih[n,k]
  k_pack_b<<<LB(6144)>>>(gru_w_hh,  bpH, 64, 192, 1,  64);  // B[k,n]=w_hh[n,k]

  // ---- 3 message-passing iterations ----
  for (int it = 0; it < 3; ++it) {
    k_zero<<<LB(NNODES * 64)>>>(agg, (long long)NNODES * 64);
    k_edge_msg<<<dim3(NEBLK), dim3(256), 0, stream>>>(hedge, out16, src, bpE, bp2, msg);
    k_scatter<<<LB(NEDGES * 64)>>>(msg, dstE, agg);
    // m = relu(out@conv_root + agg*inv_deg + conv_bias)
    {
      int waves = (NNODES / 16) * (64 / 16);
      k_gemm<<<dim3((waves + 7) / 8), dim3(256), 0, stream>>>(
          out16, bpR, conv_bias, agg, invd, mbuf, m16, NNODES, 64, 64, 1);
    }
    {
      int waves = (NNODES / 16) * (192 / 16);
      k_gemm<<<dim3((waves + 7) / 8), dim3(256), 0, stream>>>(
          m16, bpI, gru_b_ih, nullptr, nullptr, g1, nullptr, NNODES, 192, 64, 0);
      k_gemm<<<dim3((waves + 7) / 8), dim3(256), 0, stream>>>(
          out16, bpH, gru_b_hh, nullptr, nullptr, g2, nullptr, NNODES, 192, 64, 0);
    }
    k_gru<<<LB(NNODES * 64)>>>(g1, g2, out, out16);
  }

  // ---- Set2Set pooling (3 steps) ----
  k_zero<<<LB(NGRAPH * 64)>>>(hh0, (long long)NGRAPH * 64);
  k_zero<<<LB(NGRAPH * 64)>>>(ccb, (long long)NGRAPH * 64);
  k_zero<<<LB(NGRAPH * 128)>>>(qst, (long long)NGRAPH * 128);
  float* hp = hh0;
  float* hn = hh1;
  for (int s = 0; s < 3; ++s) {
    k_lstm<<<LB(NGRAPH * 64)>>>(qst, hp, ccb, hn, s2s_w_ih, s2s_w_hh, s2s_b_ih, s2s_b_hh);
    k_attdot<<<LB(NNODES)>>>(out, hn, batch, eatt);
    k_zero<<<LB(NGRAPH)>>>((float*)mxb, NGRAPH);   // 0 == encoded -inf lower bound
    k_zero<<<LB(NGRAPH)>>>(den, NGRAPH);
    k_zero<<<LB(NGRAPH * 64)>>>(rbuf, (long long)NGRAPH * 64);
    k_segmax<<<LB(NNODES)>>>(eatt, batch, mxb);
    k_atta<<<LB(NNODES)>>>(eatt, batch, mxb, aatt, den);
    k_attw<<<LB(NNODES)>>>(aatt, batch, den, watt);
    k_attr<<<LB(NNODES * 64)>>>(watt, out, batch, rbuf);
    k_qstar<<<LB(NGRAPH * 128)>>>(hn, rbuf, qst);
    float* t = hp; hp = hn; hn = t;
  }

  k_final<<<LB(NGRAPH)>>>(qst, lin1_w, lin1_b, lin2_w, lin2_b, (float*)d_out);
}
